// CLAPP_RSNN_15659450761257
// MI455X (gfx1250) — matrix operand order repaired and checked by reading, not compiled, not run on hardware
//
#include <hip/hip_runtime.h>
#include <hip/hip_bf16.h>
#include <stdint.h>

typedef _Float16 f16;
typedef __attribute__((ext_vector_type(16))) _Float16 v16h;
typedef __attribute__((ext_vector_type(8)))  float    v8f;

#define TSTEPS 100
#define BATCH  512
#define NIN    700
#define NINP   704          // inp K padded to multiple of 32
#define HID    512
#define NOUTP  20
#define BM     16           // batch rows per workgroup
#define K1PAD  1216         // NINP + HID
#define BETA   0.96f
#define NTHREADS 512        // 16 wave32s; each wave owns 32 hidden cols (2 N-tiles)

// d_out layout (floats): spk1_t[100,512,512] | spk2_t[100,512,512] | spko_t[100,512,20] | traces[2,512,512] | losses[100,2]
#define O_SPK1 ((size_t)0)
#define O_SPK2 ((size_t)26214400)
#define O_SPKO ((size_t)52428800)
#define O_TR   ((size_t)53452800)
#define O_LOSS ((size_t)53977088)

// d_ws layout (halves): W1h[512][1216] | W2h[512][1024] | Wouth[32][512]
#define WSH_W1  ((size_t)0)
#define WSH_W2  ((size_t)622592)
#define WSH_WO  ((size_t)1146880)
#define WSH_TOT ((size_t)1163264)

// ---------- prologue: fp32 weights -> f16 with concat-boundary padding ----------
__global__ void k_convert(const float* __restrict__ W1, const float* __restrict__ W2,
                          const float* __restrict__ Wout, f16* __restrict__ wsh) {
  size_t i = (size_t)blockIdx.x * 256 + threadIdx.x;
  if (i >= WSH_TOT) return;
  float v;
  if (i < WSH_W2) {                       // W1: [n][k], k<700 = input, 704.. = recurrent
    size_t n = i / K1PAD, k = i % K1PAD;
    v = (k < NIN) ? W1[n * 1212 + k] : ((k < NINP) ? 0.f : ((k - 4) < 1212 ? W1[n * 1212 + (k - 4)] : 0.f));
  } else if (i < WSH_WO) {                // W2: straight copy [512][1024]
    v = W2[i - WSH_W2];
  } else {                                // Wout: pad rows 20..31 with zeros
    size_t j = i - WSH_WO, n = j >> 9, k = j & 511;
    v = (n < NOUTP) ? Wout[n * HID + k] : 0.f;
  }
  wsh[i] = (f16)v;
}

__global__ void k_zero_loss(float* __restrict__ out) {
  int i = threadIdx.x;
  if (i < TSTEPS * 2) out[O_LOSS + i] = 0.f;
}

// ---------- fragment load helpers (contiguous k = kb + 16*g + s, same order in A and B) ----------
#define LOAD_A(dst, SBUF, kb)                                              \
  { union { uint4 u4[2]; v16h v; } _ua;                                    \
    _ua.u4[0] = *(const uint4*)&SBUF[ln][(kb) + g * 16];                   \
    _ua.u4[1] = *(const uint4*)&SBUF[ln][(kb) + 8 + g * 16];               \
    dst = _ua.v; }

#define LOAD_B(dst, WPTR)                                                  \
  { union { uint4 u4[2]; v16h v; } _ub;                                    \
    const uint4* _wp = (const uint4*)(WPTR);                               \
    _ub.u4[0] = _wp[0]; _ub.u4[1] = _wp[1];                                \
    dst = _ub.v; }

#define WMMA(acc, a, b)                                                    \
  acc = __builtin_amdgcn_wmma_f32_16x16x32_f16(false, a, false, b, (short)0, acc, false, false)

// ---------- main persistent kernel: 32 WGs x (16 batch rows), 16 waves own 32 cols each ----------
__launch_bounds__(NTHREADS)
__global__ void k_clapp(const float* __restrict__ inp,
                        const float* __restrict__ pt1,
                        const float* __restrict__ pt2,
                        const int*   __restrict__ bfp,
                        const f16*   __restrict__ W1h,
                        const f16*   __restrict__ W2h,
                        const f16*   __restrict__ Woh,
                        float* __restrict__ out) {
  __shared__ f16 s_inp[BM][NINP];   // 22.0 KB  current-step input spikes (f16)
  __shared__ f16 s_h1[BM][HID];     // 16 KB    recurrent spk1 (becomes layer-2 input)
  __shared__ f16 s_h2[BM][HID];     // 16 KB    recurrent spk2 (becomes output-layer input)
  __shared__ float s_rsum1[BM], s_rsum2[BM];
  __shared__ float s_loss[2];

  const int tid  = threadIdx.x;
  const int wave = tid >> 5, lane = tid & 31;
  const int g = lane >> 4, ln = lane & 15;   // C-layout: N = nb + ln, M = g*8 + r
  const int rowBase = blockIdx.x * BM;
  const int nb = wave * 32;                  // this wave owns hidden cols [nb, nb+32)

  // ---- init LDS state ----
  for (int i = tid; i < BM * HID; i += NTHREADS) {
    int r = i >> 9, c = i & 511;
    s_h1[r][c] = (f16)0.f; s_h2[r][c] = (f16)0.f;
  }
  if (tid < BM) { s_rsum1[tid] = 0.f; s_rsum2[tid] = 0.f; }
  if (tid < 2)  s_loss[tid] = 0.f;
  __syncthreads();

  // ---- CLAPP feedback: fb = prev_trace - row_mean ----
  { int r = tid >> 5, c0 = (tid & 31) * 16; float a = 0.f, b = 0.f;
    for (int c = 0; c < 16; ++c) {
      a += pt1[(size_t)(rowBase + r) * HID + c0 + c];
      b += pt2[(size_t)(rowBase + r) * HID + c0 + c];
    }
    atomicAdd(&s_rsum1[r], a); atomicAdd(&s_rsum2[r], b); }
  __syncthreads();

  v8f fb1[2], fb2[2], m1[2], m2[2], tr1[2], tr2[2];
#pragma unroll
  for (int q = 0; q < 2; ++q) {
#pragma unroll
    for (int r = 0; r < 8; ++r) {
      int lr = g * 8 + r; int br = rowBase + lr; int n = nb + q * 16 + ln;
      fb1[q][r] = pt1[(size_t)br * HID + n] - s_rsum1[lr] * (1.f / HID);
      fb2[q][r] = pt2[(size_t)br * HID + n] - s_rsum2[lr] * (1.f / HID);
      m1[q][r] = 0.f; m2[q][r] = 0.f; tr1[q][r] = 0.f; tr2[q][r] = 0.f;
    }
  }
  v8f mo[2];
#pragma unroll
  for (int q = 0; q < 2; ++q)
#pragma unroll
    for (int r = 0; r < 8; ++r) mo[q][r] = 0.f;
  const float nbf = -(float)bfp[0];

  for (int t = 0; t < TSTEPS; ++t) {
    // stage input spikes (f32 -> f16, pad cols 700..703)
    const float* ip = inp + ((size_t)t * BATCH + rowBase) * NIN;
#pragma clang loop unroll_count(4)
    for (int i = tid; i < BM * NINP; i += NTHREADS) {
      int r = i / NINP, c = i - r * NINP;
      s_inp[r][c] = (f16)((c < NIN) ? ip[(size_t)r * NIN + c] : 0.f);
    }
    __syncthreads();

    // ---------------- layer 1: cur1 = [inp | h1] @ W1^T ----------------
    v8f acc[2];
#pragma unroll
    for (int q = 0; q < 2; ++q)
#pragma unroll
      for (int r = 0; r < 8; ++r) acc[q][r] = 0.f;
#pragma clang loop unroll_count(2)
    for (int kb = 0; kb < NINP; kb += 32) {
      v16h a; LOAD_A(a, s_inp, kb);
#pragma unroll
      for (int q = 0; q < 2; ++q) {
        v16h b; LOAD_B(b, W1h + (size_t)(nb + q * 16 + ln) * K1PAD + kb + g * 16);
        WMMA(acc[q], a, b);
      }
    }
#pragma clang loop unroll_count(2)
    for (int kb = 0; kb < HID; kb += 32) {
      v16h a; LOAD_A(a, s_h1, kb);
#pragma unroll
      for (int q = 0; q < 2; ++q) {
        v16h b; LOAD_B(b, W1h + (size_t)(nb + q * 16 + ln) * K1PAD + NINP + kb + g * 16);
        WMMA(acc[q], a, b);
      }
    }
    // LIF1 + trace + loss partial + global spike store
    float lp1 = 0.f;
#pragma unroll
    for (int q = 0; q < 2; ++q) {
#pragma unroll
      for (int r = 0; r < 8; ++r) {
        float m = m1[q][r];
        float rst = (m > 1.f) ? 1.f : 0.f;
        m = BETA * m + acc[q][r] - rst;
        m1[q][r] = m;
        float s = (m > 1.f) ? 1.f : 0.f;
        tr1[q][r] = (t == 0) ? s : tr1[q][r] + s * (1.f / TSTEPS);
        lp1 += s * fb1[q][r];
        out[O_SPK1 + ((size_t)t * BATCH + rowBase + g * 8 + r) * HID + nb + q * 16 + ln] = s;
      }
    }
#pragma unroll
    for (int off = 16; off; off >>= 1) lp1 += __shfl_xor(lp1, off, 32);
    __syncthreads();                                   // all waves done reading s_h1
#pragma unroll
    for (int q = 0; q < 2; ++q)
#pragma unroll
      for (int r = 0; r < 8; ++r)
        s_h1[g * 8 + r][nb + q * 16 + ln] = (f16)((m1[q][r] > 1.f) ? 1.f : 0.f);
    if (lane == 0) atomicAdd(&s_loss[0], lp1);
    __syncthreads();                                   // spk1 published

    // ---------------- layer 2: cur2 = [spk1 | h2] @ W2^T ----------------
#pragma unroll
    for (int q = 0; q < 2; ++q)
#pragma unroll
      for (int r = 0; r < 8; ++r) acc[q][r] = 0.f;
#pragma clang loop unroll_count(2)
    for (int kb = 0; kb < HID; kb += 32) {
      v16h a; LOAD_A(a, s_h1, kb);
#pragma unroll
      for (int q = 0; q < 2; ++q) {
        v16h b; LOAD_B(b, W2h + (size_t)(nb + q * 16 + ln) * (2 * HID) + kb + g * 16);
        WMMA(acc[q], a, b);
      }
    }
#pragma clang loop unroll_count(2)
    for (int kb = 0; kb < HID; kb += 32) {
      v16h a; LOAD_A(a, s_h2, kb);
#pragma unroll
      for (int q = 0; q < 2; ++q) {
        v16h b; LOAD_B(b, W2h + (size_t)(nb + q * 16 + ln) * (2 * HID) + HID + kb + g * 16);
        WMMA(acc[q], a, b);
      }
    }
    float lp2 = 0.f;
#pragma unroll
    for (int q = 0; q < 2; ++q) {
#pragma unroll
      for (int r = 0; r < 8; ++r) {
        float m = m2[q][r];
        float rst = (m > 1.f) ? 1.f : 0.f;
        m = BETA * m + acc[q][r] - rst;
        m2[q][r] = m;
        float s = (m > 1.f) ? 1.f : 0.f;
        tr2[q][r] = (t == 0) ? s : tr2[q][r] + s * (1.f / TSTEPS);
        lp2 += s * fb2[q][r];
        out[O_SPK2 + ((size_t)t * BATCH + rowBase + g * 8 + r) * HID + nb + q * 16 + ln] = s;
      }
    }
#pragma unroll
    for (int off = 16; off; off >>= 1) lp2 += __shfl_xor(lp2, off, 32);
    __syncthreads();                                   // all waves done reading s_h2
#pragma unroll
    for (int q = 0; q < 2; ++q)
#pragma unroll
      for (int r = 0; r < 8; ++r)
        s_h2[g * 8 + r][nb + q * 16 + ln] = (f16)((m2[q][r] > 1.f) ? 1.f : 0.f);
    if (lane == 0) atomicAdd(&s_loss[1], lp2);
    __syncthreads();                                   // spk2 published

    // ---------------- output layer (wave 0 only, 20 cols padded to 32) ----------------
    if (wave == 0) {
      v8f ao[2];
#pragma unroll
      for (int q = 0; q < 2; ++q)
#pragma unroll
        for (int r = 0; r < 8; ++r) ao[q][r] = 0.f;
#pragma clang loop unroll(disable)
      for (int kb = 0; kb < HID; kb += 32) {
        v16h a; LOAD_A(a, s_h2, kb);
#pragma unroll
        for (int q = 0; q < 2; ++q) {
          v16h b; LOAD_B(b, Woh + (size_t)(q * 16 + ln) * HID + kb + g * 16);
          WMMA(ao[q], a, b);
        }
      }
#pragma unroll
      for (int q = 0; q < 2; ++q) {
#pragma unroll
        for (int r = 0; r < 8; ++r) {
          float m = mo[q][r];
          float rst = (m > 1.f) ? 1.f : 0.f;
          m = BETA * m + ao[q][r] - rst;
          mo[q][r] = m;
          float s = (m > 1.f) ? 1.f : 0.f;
          int n = q * 16 + ln;
          if (n < NOUTP)
            out[O_SPKO + ((size_t)t * BATCH + rowBase + g * 8 + r) * NOUTP + n] = s;
        }
      }
    }
    if (tid == 0) {
      atomicAdd(&out[O_LOSS + 2 * t + 0], s_loss[0] * nbf * (1.f / BATCH));
      atomicAdd(&out[O_LOSS + 2 * t + 1], s_loss[1] * nbf * (1.f / BATCH));
      s_loss[0] = 0.f; s_loss[1] = 0.f;
    }
    __syncthreads();                                   // end of step
  }

  // ---- final traces [2, B, H] ----
#pragma unroll
  for (int q = 0; q < 2; ++q) {
#pragma unroll
    for (int r = 0; r < 8; ++r) {
      size_t br = (size_t)rowBase + g * 8 + r; int n = nb + q * 16 + ln;
      out[O_TR + br * HID + n] = tr1[q][r];
      out[O_TR + (size_t)BATCH * HID + br * HID + n] = tr2[q][r];
    }
  }
}

extern "C" void kernel_launch(void* const* d_in, const int* in_sizes, int n_in,
                              void* d_out, int out_size, void* d_ws, size_t ws_size,
                              hipStream_t stream) {
  (void)in_sizes; (void)n_in; (void)out_size; (void)ws_size;
  const float* inp  = (const float*)d_in[0];
  const float* W1   = (const float*)d_in[1];
  const float* W2   = (const float*)d_in[2];
  const float* Wout = (const float*)d_in[3];
  const float* pt1  = (const float*)d_in[4];
  const float* pt2  = (const float*)d_in[5];
  const int*   bf   = (const int*)d_in[6];
  float* out = (float*)d_out;
  f16*   wsh = (f16*)d_ws;

  k_convert<<<(unsigned)((WSH_TOT + 255) / 256), 256, 0, stream>>>(W1, W2, Wout, wsh);
  k_zero_loss<<<1, 256, 0, stream>>>(out);
  k_clapp<<<BATCH / BM, NTHREADS, 0, stream>>>(inp, pt1, pt2, bf,
                                               wsh + WSH_W1, wsh + WSH_W2, wsh + WSH_WO, out);
}
